// ConceptBank_83588653515221
// MI455X (gfx1250) — matrix operand order repaired and checked by reading, not compiled, not run on hardware
//
#include <hip/hip_runtime.h>
#include <hip/hip_bf16.h>
#include <math.h>

// Problem constants (from reference): B=64, N=4096, K=64, D=256, n_slots=8
#define CB_B   64
#define CB_N   4096
#define CB_K   64
#define CB_D   256
#define CB_S   8

typedef __bf16 v16bf __attribute__((ext_vector_type(16)));
typedef float  v8f   __attribute__((ext_vector_type(8)));

// ---------------------------------------------------------------------------
// Kernel 1: normalize mu rows -> bf16 (mn), zero the per-batch concept sums.
// ---------------------------------------------------------------------------
__global__ __launch_bounds__(256) void cb_prep(const float* __restrict__ mu,
                                               __bf16* __restrict__ mn,
                                               float* __restrict__ r_sum) {
  const int tid = threadIdx.x;
  for (int i = tid; i < CB_B * CB_K; i += 256) r_sum[i] = 0.0f;

  const int wave = tid >> 5;
  const int lane = tid & 31;
  for (int row = wave; row < CB_K; row += 8) {
    const float* src = mu + (size_t)row * CB_D;
    float vals[8];
    float s = 0.0f;
#pragma unroll
    for (int j = 0; j < 8; ++j) {
      vals[j] = src[lane + 32 * j];
      s += vals[j] * vals[j];
    }
#pragma unroll
    for (int off = 16; off >= 1; off >>= 1) s += __shfl_xor(s, off, 32);
    const float rn = rsqrtf(fmaxf(s, 1e-16f));  // == 1/max(||mu||,1e-8)
    __bf16* dst = mn + (size_t)row * CB_D;
#pragma unroll
    for (int j = 0; j < 8; ++j) dst[lane + 32 * j] = (__bf16)(vals[j] * rn);
  }
}

// ---------------------------------------------------------------------------
// Kernel 2: cos = (x/||x||) . mn^T via v_wmma_f32_16x16x32_bf16, fused
// softmax over K=64 per row, accumulate sum over rows into r_sum[b,k].
// Grid: B * (N/128) blocks of 256 threads (8 waves x 16 rows).
// ---------------------------------------------------------------------------
__global__ __launch_bounds__(256) void cb_main(const float* __restrict__ x,
                                               const __bf16* __restrict__ mn,
                                               float* __restrict__ r_sum) {
  const int b    = blockIdx.x >> 5;   // 32 row-tiles per batch
  const int tile = blockIdx.x & 31;
  const int wave = threadIdx.x >> 5;
  const int lane = threadIdx.x & 31;
  const int row0 = tile * 128 + wave * 16;  // first row (within batch) of this wave

  __shared__ float s_rsum[CB_K];
  if (threadIdx.x < CB_K) s_rsum[threadIdx.x] = 0.0f;
  __syncthreads();

  const float* xtile = x + ((size_t)b * CB_N + row0) * CB_D;

  // --- pass 1: per-row 1/||x|| (16 rows per wave; reread hits L2) ---------
  float rnorm = 0.0f;
  for (int j = 0; j < 16; ++j) {
    const float* src = xtile + (size_t)j * CB_D;
    float s = 0.0f;
#pragma unroll
    for (int t = 0; t < 8; ++t) {
      const float v = src[lane + 32 * t];
      s += v * v;
    }
#pragma unroll
    for (int off = 16; off >= 1; off >>= 1) s += __shfl_xor(s, off, 32);
    const float rn = rsqrtf(fmaxf(s, 1e-16f));
    if ((lane & 15) == j) rnorm = rn;  // lane L and L+16 own row (L&15)
  }

  // --- pass 2: WMMA GEMM, C[16 rows x 64 concepts] per wave ---------------
  // A 16x32 bf16 layout: lanes 0-15 row M=lane hold K d0..d0+15,
  // lanes 16-31 row M=lane-16 hold K d0+16..d0+31.
  const float* arow  = x + ((size_t)b * CB_N + row0 + (lane & 15)) * CB_D;
  const int    choff = (lane < 16) ? 0 : 16;

  v8f acc[4];
#pragma unroll
  for (int kt = 0; kt < 4; ++kt)
#pragma unroll
    for (int v = 0; v < 8; ++v) acc[kt][v] = 0.0f;

#pragma unroll
  for (int s = 0; s < 8; ++s) {              // D = 8 steps of 32
    const float* p0 = arow + s * 32 + choff; // 32B-aligned
    const float* p1 = p0 + 8;
    const v8f c0 = *(const v8f*)p0;
    const v8f c1 = *(const v8f*)p1;
    v16bf a;
#pragma unroll
    for (int t = 0; t < 8; ++t) {
      a[t]     = (__bf16)(c0[t] * rnorm);
      a[8 + t] = (__bf16)(c1[t] * rnorm);
    }
#pragma unroll
    for (int kt = 0; kt < 4; ++kt) {
      // B 32x16 bf16: lane column = lane&15; lanes<16 hold K d0..d0+15,
      // lanes>=16 hold K d0+16..d0+31 -> one contiguous 32B load from mn.
      const __bf16* bp =
          mn + (size_t)(kt * 16 + (lane & 15)) * CB_D + s * 32 + ((lane < 16) ? 0 : 16);
      const v16bf bfrag = *(const v16bf*)bp;
      acc[kt] = __builtin_amdgcn_wmma_f32_16x16x32_bf16(
          false, a, false, bfrag, (short)0, acc[kt], false, false);
    }
  }

  // --- fused softmax over K=64 per row, summed over the wave's 16 rows ----
  // C layout: acc[kt][v] = cos(row M = v + (lane<16?0:8), concept kt*16+(lane&15)).
  float psum[4] = {0.0f, 0.0f, 0.0f, 0.0f};
#pragma unroll
  for (int v = 0; v < 8; ++v) {
    float m = fmaxf(fmaxf(acc[0][v], acc[1][v]), fmaxf(acc[2][v], acc[3][v]));
#pragma unroll
    for (int off = 8; off >= 1; off >>= 1) m = fmaxf(m, __shfl_xor(m, off, 32));
    const float e0 = __expf(acc[0][v] - m);
    const float e1 = __expf(acc[1][v] - m);
    const float e2 = __expf(acc[2][v] - m);
    const float e3 = __expf(acc[3][v] - m);
    float s = e0 + e1 + e2 + e3;
#pragma unroll
    for (int off = 8; off >= 1; off >>= 1) s += __shfl_xor(s, off, 32);
    const float inv = 1.0f / s;
    psum[0] += e0 * inv;
    psum[1] += e1 * inv;
    psum[2] += e2 * inv;
    psum[3] += e3 * inv;
  }
#pragma unroll
  for (int kt = 0; kt < 4; ++kt)
    atomicAdd(&s_rsum[kt * 16 + (lane & 15)], psum[kt]);  // ds_add_f32
  __syncthreads();
  if (threadIdx.x < CB_K)
    atomicAdd(&r_sum[b * CB_K + threadIdx.x], s_rsum[threadIdx.x]);
}

// ---------------------------------------------------------------------------
// Kernel 3: top-8 (softmax before top_k is monotonic -> skip it), then
// out = mu[idx] + exp(log_sigma[idx]) * noise.
// ---------------------------------------------------------------------------
__device__ __forceinline__ unsigned cb_pcg(unsigned v) {
  unsigned s = v * 747796405u + 2891336453u;
  unsigned w = ((s >> ((s >> 28) + 4u)) ^ s) * 277803737u;
  return (w >> 22) ^ w;
}

__device__ __forceinline__ float cb_gauss(unsigned idx) {
  const unsigned a = cb_pcg(idx * 2u + 42u);
  const unsigned b = cb_pcg(idx * 2u + 1101u);
  const float u1 = fmaxf((float)a * 2.3283064e-10f, 1e-12f);
  const float u2 = (float)b * 2.3283064e-10f;
  return sqrtf(-2.0f * __logf(u1)) * __cosf(6.2831853f * u2);
}

__global__ __launch_bounds__(256) void cb_out(const float* __restrict__ r_sum,
                                              const float* __restrict__ mu,
                                              const float* __restrict__ log_sigma,
                                              float* __restrict__ out) {
  const int b = blockIdx.x;
  __shared__ int s_idx[CB_S];
  const int tid = threadIdx.x;

  if (tid < 32) {
    float v0 = r_sum[b * CB_K + tid];
    float v1 = r_sum[b * CB_K + 32 + tid];
    int i0 = tid, i1 = 32 + tid;
    for (int slot = 0; slot < CB_S; ++slot) {
      float v = v0;
      int   i = i0;
      if (v1 > v || (v1 == v && i1 < i)) { v = v1; i = i1; }
#pragma unroll
      for (int off = 16; off >= 1; off >>= 1) {
        const float ov = __shfl_xor(v, off, 32);
        const int   oi = __shfl_xor(i, off, 32);
        if (ov > v || (ov == v && oi < i)) { v = ov; i = oi; }
      }
      if (tid == 0) s_idx[slot] = i;
      if (i0 == i) v0 = -3.4e38f;
      if (i1 == i) v1 = -3.4e38f;
    }
  }
  __syncthreads();

  for (int e = tid; e < CB_S * CB_D; e += 256) {
    const int slot = e >> 8;
    const int d    = e & (CB_D - 1);
    const int k    = s_idx[slot];
    const float m  = mu[k * CB_D + d];
    const float sg = __expf(log_sigma[k * CB_D + d]);
    const unsigned gidx = (unsigned)(((b * CB_S) + slot) * CB_D + d);
    out[(size_t)b * (CB_S * CB_D) + e] = m + sg * cb_gauss(gidx);
  }
}

// ---------------------------------------------------------------------------
extern "C" void kernel_launch(void* const* d_in, const int* in_sizes, int n_in,
                              void* d_out, int out_size, void* d_ws, size_t ws_size,
                              hipStream_t stream) {
  (void)in_sizes; (void)n_in; (void)out_size; (void)ws_size;
  const float* x         = (const float*)d_in[0];
  const float* mu        = (const float*)d_in[1];
  const float* log_sigma = (const float*)d_in[2];
  // d_in[3] = n_slots (reference fixes it at 8; baked into CB_S)
  float* out = (float*)d_out;

  __bf16* mn    = (__bf16*)d_ws;                                    // 32 KB
  float*  r_sum = (float*)((char*)d_ws + (size_t)CB_K * CB_D * 2);  // 16 KB

  cb_prep<<<1, 256, 0, stream>>>(mu, mn, r_sum);
  cb_main<<<CB_B * (CB_N / 128), 256, 0, stream>>>(x, mn, r_sum);
  cb_out<<<CB_B, 256, 0, stream>>>(r_sum, mu, log_sigma, out);
}
